// MoE_47476568490356
// MI455X (gfx1250) — compile-verified, matrix operand order checked by reading, not run on hardware
//
#include <hip/hip_runtime.h>

// Problem constants (match reference)
#define TOK 2048      // B*S tokens
#define HD  2048      // hidden
#define NE  8         // routed experts
#define KK  2         // top-k
#define FD  1408      // routed intermediate
#define FSD 2816      // shared intermediate (2*F)

typedef __attribute__((ext_vector_type(16))) _Float16 v16h;
typedef __attribute__((ext_vector_type(8)))  float    v8f;
typedef int v4i_vec __attribute__((vector_size(16)));   // async-to-LDS builtin operand type
typedef __attribute__((address_space(3))) v4i_vec* lds_v4i_p;

union HFrag { v16h v; unsigned u[8]; };
union H8    { _Float16 h[8]; uint2 u2[2]; };

// LDS tile pitch: 32 K-halfs + 8 pad = 80B rows (16B-aligned rows for async b128,
// conflict-free fragment reads).
#define PITCH 40

#if __has_builtin(__builtin_amdgcn_global_load_async_to_lds_b128)
#define HAVE_ASYNC_LDS 1
#endif

__device__ __forceinline__ v8f vzero() {
  v8f z;
#pragma unroll
  for (int i = 0; i < 8; ++i) z[i] = 0.f;
  return z;
}

__device__ __forceinline__ v8f wmma16(v16h a, v16h b, v8f c) {
  return __builtin_amdgcn_wmma_f32_16x16x32_f16(false, a, false, b, (short)0, c, false, false);
}

// CDNA5 16-bit A/B fragment layout: lane&15 = row; lanes 0-15 hold K {0..7,16..23},
// lanes 16-31 hold K {8..15,24..31}; each VGPR packs 2 consecutive K halfs.
__device__ __forceinline__ v16h ldsFrag(const _Float16* lds, int rowBase, int lane) {
  const int row = rowBase + (lane & 15);
  const unsigned* p = (const unsigned*)(lds + row * PITCH);
  const int hi = (lane < 16) ? 0 : 4;
  HFrag f;
#pragma unroll
  for (int v = 0; v < 8; ++v) {
    const int d = ((v < 4) ? v : (v + 4)) + hi;
    f.u[v] = p[d];
  }
  return f.v;
}

// 32B (16 halfs) global -> LDS, async DMA on gfx1250 (ASYNCcnt), else sync copy.
__device__ __forceinline__ void issue32B(const _Float16* g, _Float16* d) {
#if defined(HAVE_ASYNC_LDS)
  __builtin_amdgcn_global_load_async_to_lds_b128((v4i_vec*)g, (lds_v4i_p)d, 0, 0);
  __builtin_amdgcn_global_load_async_to_lds_b128((v4i_vec*)(g + 8), (lds_v4i_p)(d + 8), 0, 0);
#else
  const uint2* gg = (const uint2*)g;
  uint2* dd = (uint2*)d;
#pragma unroll
  for (int i = 0; i < 4; ++i) dd[i] = gg[i];
#endif
}

// 128x32 f16 A-tile from row-contiguous source.
__device__ __forceinline__ void tileA_issue(_Float16* ldsDst, const _Float16* src,
                                            int row0, int ld, int k0, int tid) {
  const int row  = tid >> 1;
  const int part = (tid & 1) * 16;
  issue32B(src + (size_t)(row0 + row) * ld + k0 + part, ldsDst + row * PITCH + part);
}

__device__ __forceinline__ void waitAsync() {
#if defined(HAVE_ASYNC_LDS)
#if __has_builtin(__builtin_amdgcn_s_wait_asynccnt)
  __builtin_amdgcn_s_wait_asynccnt(0);
#else
  asm volatile("s_wait_asynccnt 0x0" ::: "memory");
#endif
#endif
}

// Register staging for one 64x32 fp32 weight tile slice (8 floats/thread).
struct BReg { float4 f0, f1; };

__device__ __forceinline__ void prefB(BReg& r, const float* src, int row0, int ld,
                                      int k0, int tid) {
  const int row = tid >> 2;
  const int c8  = (tid & 3) * 8;
  const float4* g = (const float4*)(src + (size_t)(row0 + row) * ld + k0 + c8);
  r.f0 = g[0]; r.f1 = g[1];
}

__device__ __forceinline__ void storeB(_Float16* lds, const BReg& r, int tid) {
  const int row = tid >> 2;
  const int c8  = (tid & 3) * 8;
  H8 h;
  h.h[0] = (_Float16)r.f0.x; h.h[1] = (_Float16)r.f0.y;
  h.h[2] = (_Float16)r.f0.z; h.h[3] = (_Float16)r.f0.w;
  h.h[4] = (_Float16)r.f1.x; h.h[5] = (_Float16)r.f1.y;
  h.h[6] = (_Float16)r.f1.z; h.h[7] = (_Float16)r.f1.w;
  uint2* d = (uint2*)(lds + row * PITCH + c8);
  d[0] = h.u2[0]; d[1] = h.u2[1];
}

__device__ __forceinline__ float silu(float g) {
  return g * __builtin_amdgcn_rcpf(1.f + __expf(-g));   // v_rcp_f32, no IEEE-div expansion
}

// -------- K1: gating (softmax + greedy top-2), expert lists, X fp32->f16 --------
__global__ void __launch_bounds__(256) gate_kernel(const float* __restrict__ X,
                                                   const float* __restrict__ GW,
                                                   _Float16* __restrict__ Xh,
                                                   float* __restrict__ w,
                                                   int* __restrict__ cnt,
                                                   int* __restrict__ idxSlot) {
  const int lane = threadIdx.x & 31;
  const int t = blockIdx.x * 8 + (threadIdx.x >> 5);
  float acc[NE];
#pragma unroll
  for (int e = 0; e < NE; ++e) acc[e] = 0.f;
  const float* xrow = X + (size_t)t * HD;
  for (int h0 = 0; h0 < HD; h0 += 32) {
    const float xv = xrow[h0 + lane];
    Xh[(size_t)t * HD + h0 + lane] = (_Float16)xv;
#pragma unroll
    for (int e = 0; e < NE; ++e) acc[e] += xv * GW[e * HD + h0 + lane];
  }
#pragma unroll
  for (int off = 16; off > 0; off >>= 1) {
#pragma unroll
    for (int e = 0; e < NE; ++e) acc[e] += __shfl_xor(acc[e], off, 32);
  }
  if (lane == 0) {
    float m = acc[0];
#pragma unroll
    for (int e = 1; e < NE; ++e) m = fmaxf(m, acc[e]);
    float p[NE]; float s = 0.f;
#pragma unroll
    for (int e = 0; e < NE; ++e) { p[e] = __expf(acc[e] - m); s += p[e]; }
    const float inv = 1.f / s;
#pragma unroll
    for (int e = 0; e < NE; ++e) p[e] *= inv;
    int i1 = 0;
#pragma unroll
    for (int e = 1; e < NE; ++e) if (p[e] > p[i1]) i1 = e; // ties -> lowest idx
    int i2 = (i1 == 0) ? 1 : 0;
#pragma unroll
    for (int e = 0; e < NE; ++e) if (e != i1 && e != i2 && p[e] > p[i2]) i2 = e;
    const float sw = p[i1] + p[i2] + 1e-20f;
    w[t * KK + 0] = p[i1] / sw;       // * ROUTED_SCALE (=1.0)
    w[t * KK + 1] = p[i2] / sw;
    const int pos0 = atomicAdd(&cnt[i1], 1);
    idxSlot[i1 * TOK + pos0] = t * KK + 0;
    const int pos1 = atomicAdd(&cnt[i2], 1);
    idxSlot[i2 * TOK + pos1] = t * KK + 1;
  }
}

// -------- shared-expert gate&up GEMMs + SiLU*up (dense over all tokens) --------
__global__ void __launch_bounds__(256) upshared_kernel(const _Float16* __restrict__ Xh,
                                                       const float* __restrict__ Wg,
                                                       const float* __restrict__ Wu,
                                                       _Float16* __restrict__ Hout) {
  __shared__ __align__(16) _Float16 sA [2][128 * PITCH];
  __shared__ __align__(16) _Float16 sBg[2][64 * PITCH];
  __shared__ __align__(16) _Float16 sBu[2][64 * PITCH];
  const int tid  = threadIdx.x;
  const int lane = tid & 31;
  const int wave = tid >> 5;
  const int wm = (wave & 3) * 32;
  const int wn = (wave >> 2) * 32;
  const int t0 = blockIdx.x * 128;
  const int n0 = blockIdx.y * 64;

  v8f cg[2][2], cu[2][2];
#pragma unroll
  for (int i = 0; i < 2; ++i)
#pragma unroll
    for (int j = 0; j < 2; ++j) { cg[i][j] = vzero(); cu[i][j] = vzero(); }

  BReg bg, bu;
  prefB(bg, Wg, n0, HD, 0, tid);
  prefB(bu, Wu, n0, HD, 0, tid);
  tileA_issue(sA[0], Xh, t0, HD, 0, tid);
  storeB(sBg[0], bg, tid);
  storeB(sBu[0], bu, tid);
  waitAsync();
  __syncthreads();

  int cur = 0;
  for (int k0 = 0; k0 < HD; k0 += 32) {
    const int nxt = cur ^ 1;
    const bool has = (k0 + 32) < HD;
    if (has) {
      tileA_issue(sA[nxt], Xh, t0, HD, k0 + 32, tid);
      prefB(bg, Wg, n0, HD, k0 + 32, tid);
      prefB(bu, Wu, n0, HD, k0 + 32, tid);
    }
    const v16h a0 = ldsFrag(sA[cur], wm,      lane);
    const v16h a1 = ldsFrag(sA[cur], wm + 16, lane);
    const v16h g0 = ldsFrag(sBg[cur], wn,      lane);
    const v16h g1 = ldsFrag(sBg[cur], wn + 16, lane);
    const v16h u0 = ldsFrag(sBu[cur], wn,      lane);
    const v16h u1 = ldsFrag(sBu[cur], wn + 16, lane);
    cg[0][0] = wmma16(a0, g0, cg[0][0]);
    cg[0][1] = wmma16(a0, g1, cg[0][1]);
    cg[1][0] = wmma16(a1, g0, cg[1][0]);
    cg[1][1] = wmma16(a1, g1, cg[1][1]);
    cu[0][0] = wmma16(a0, u0, cu[0][0]);
    cu[0][1] = wmma16(a0, u1, cu[0][1]);
    cu[1][0] = wmma16(a1, u0, cu[1][0]);
    cu[1][1] = wmma16(a1, u1, cu[1][1]);
    if (has) {
      storeB(sBg[nxt], bg, tid);
      storeB(sBu[nxt], bu, tid);
    }
    waitAsync();
    __syncthreads();
    cur = nxt;
  }

  const int rlo = (lane < 16) ? 0 : 8;
  const int cn  = lane & 15;
#pragma unroll
  for (int i = 0; i < 2; ++i)
#pragma unroll
    for (int j = 0; j < 2; ++j)
#pragma unroll
      for (int r = 0; r < 8; ++r) {
        const float a = silu(cg[i][j][r]) * cu[i][j][r];
        const int t = t0 + wm + i * 16 + r + rlo;
        const int f = n0 + wn + j * 16 + cn;
        Hout[(size_t)t * FSD + f] = (_Float16)a;
      }
}

// -------- routed gate&up GEMMs over gathered token lists --------
__global__ void __launch_bounds__(256) upgather_kernel(const _Float16* __restrict__ Xh,
                                                       const float* __restrict__ Wg,
                                                       const float* __restrict__ Wu,
                                                       _Float16* __restrict__ Hg,
                                                       const int* __restrict__ cnt,
                                                       const int* __restrict__ idxSlot) {
  const int e  = blockIdx.z;
  const int ce = cnt[e];
  const int r0 = blockIdx.x * 128;
  if (r0 >= ce) return;   // uniform early-exit: EXEC stays all-ones for WMMA blocks

  __shared__ __align__(16) _Float16 sA [2][128 * PITCH];
  __shared__ __align__(16) _Float16 sBg[2][64 * PITCH];
  __shared__ __align__(16) _Float16 sBu[2][64 * PITCH];
  const int tid  = threadIdx.x;
  const int lane = tid & 31;
  const int wave = tid >> 5;
  const int wm = (wave & 3) * 32;
  const int wn = (wave >> 2) * 32;
  const int n0 = blockIdx.y * 64;
  const float* wgB = Wg + (size_t)e * FD * HD;
  const float* wuB = Wu + (size_t)e * FD * HD;

  // Per-thread gathered A-row pointer (clamped padding rows; stores masked later).
  const int myRow = tid >> 1;
  const int part  = (tid & 1) * 16;
  int pp = r0 + myRow; if (pp > ce - 1) pp = ce - 1;
  const int tok = idxSlot[e * TOK + pp] >> 1;     // slot = t*KK + k
  const _Float16* aRow = Xh + (size_t)tok * HD + part;
  _Float16* aDst0 = sA[0] + myRow * PITCH + part;
  _Float16* aDst1 = sA[1] + myRow * PITCH + part;

  v8f cg[2][2], cu[2][2];
#pragma unroll
  for (int i = 0; i < 2; ++i)
#pragma unroll
    for (int j = 0; j < 2; ++j) { cg[i][j] = vzero(); cu[i][j] = vzero(); }

  BReg bg, bu;
  prefB(bg, wgB, n0, HD, 0, tid);
  prefB(bu, wuB, n0, HD, 0, tid);
  issue32B(aRow, aDst0);
  storeB(sBg[0], bg, tid);
  storeB(sBu[0], bu, tid);
  waitAsync();
  __syncthreads();

  int cur = 0;
  for (int k0 = 0; k0 < HD; k0 += 32) {
    const int nxt = cur ^ 1;
    const bool has = (k0 + 32) < HD;
    if (has) {
      issue32B(aRow + k0 + 32, (nxt ? aDst1 : aDst0));
      prefB(bg, wgB, n0, HD, k0 + 32, tid);
      prefB(bu, wuB, n0, HD, k0 + 32, tid);
    }
    const v16h a0 = ldsFrag(sA[cur], wm,      lane);
    const v16h a1 = ldsFrag(sA[cur], wm + 16, lane);
    const v16h g0 = ldsFrag(sBg[cur], wn,      lane);
    const v16h g1 = ldsFrag(sBg[cur], wn + 16, lane);
    const v16h u0 = ldsFrag(sBu[cur], wn,      lane);
    const v16h u1 = ldsFrag(sBu[cur], wn + 16, lane);
    cg[0][0] = wmma16(a0, g0, cg[0][0]);
    cg[0][1] = wmma16(a0, g1, cg[0][1]);
    cg[1][0] = wmma16(a1, g0, cg[1][0]);
    cg[1][1] = wmma16(a1, g1, cg[1][1]);
    cu[0][0] = wmma16(a0, u0, cu[0][0]);
    cu[0][1] = wmma16(a0, u1, cu[0][1]);
    cu[1][0] = wmma16(a1, u0, cu[1][0]);
    cu[1][1] = wmma16(a1, u1, cu[1][1]);
    if (has) {
      storeB(sBg[nxt], bg, tid);
      storeB(sBu[nxt], bu, tid);
    }
    waitAsync();
    __syncthreads();
    cur = nxt;
  }

  const int rlo = (lane < 16) ? 0 : 8;
  const int cn  = lane & 15;
#pragma unroll
  for (int i = 0; i < 2; ++i)
#pragma unroll
    for (int j = 0; j < 2; ++j)
#pragma unroll
      for (int r = 0; r < 8; ++r) {
        const int p = r0 + wm + i * 16 + r + rlo;  // position in expert list
        if (p < ce) {
          const float a = silu(cg[i][j][r]) * cu[i][j][r];
          const int f = n0 + wn + j * 16 + cn;
          Hg[((size_t)e * TOK + p) * FD + f] = (_Float16)a;
        }
      }
}

// Pipelined GEMM segment: ct += A[r0.., k]*B[n0.., k]^T over Ksz, f16 A / fp32 B.
__device__ __forceinline__ void gemm_seg(const _Float16* aSrc, int lda,
                                         const float* bSrc, int ldb, int Ksz,
                                         _Float16 (*sA)[128 * PITCH],
                                         _Float16 (*sB)[64 * PITCH],
                                         int r0, int n0, int tid, int lane,
                                         int wm, int wn, v8f ct[2][2]) {
  BReg br;
  prefB(br, bSrc, n0, ldb, 0, tid);
  tileA_issue(sA[0], aSrc, r0, lda, 0, tid);
  storeB(sB[0], br, tid);
  waitAsync();
  __syncthreads();
  int cur = 0;
  for (int k0 = 0; k0 < Ksz; k0 += 32) {
    const int nxt = cur ^ 1;
    const bool has = (k0 + 32) < Ksz;
    if (has) {
      tileA_issue(sA[nxt], aSrc, r0, lda, k0 + 32, tid);
      prefB(br, bSrc, n0, ldb, k0 + 32, tid);
    }
    const v16h a0 = ldsFrag(sA[cur], wm,      lane);
    const v16h a1 = ldsFrag(sA[cur], wm + 16, lane);
    const v16h b0 = ldsFrag(sB[cur], wn,      lane);
    const v16h b1 = ldsFrag(sB[cur], wn + 16, lane);
    ct[0][0] = wmma16(a0, b0, ct[0][0]);
    ct[0][1] = wmma16(a0, b1, ct[0][1]);
    ct[1][0] = wmma16(a1, b0, ct[1][0]);
    ct[1][1] = wmma16(a1, b1, ct[1][1]);
    if (has) storeB(sB[nxt], br, tid);
    waitAsync();
    __syncthreads();
    cur = nxt;
  }
}

// -------- routed down GEMMs over gathered lists -> (t,k)-slot outputs --------
__global__ void __launch_bounds__(256) downgather_kernel(const _Float16* __restrict__ Hg,
                                                         const float* __restrict__ Wd,
                                                         _Float16* __restrict__ Dslot,
                                                         const int* __restrict__ cnt,
                                                         const int* __restrict__ idxSlot) {
  const int e  = blockIdx.z;
  const int ce = cnt[e];
  const int r0 = blockIdx.x * 128;
  if (r0 >= ce) return;

  __shared__ __align__(16) _Float16 sA[2][128 * PITCH];
  __shared__ __align__(16) _Float16 sB[2][64 * PITCH];
  __shared__ int sSlot[128];
  const int tid  = threadIdx.x;
  const int lane = tid & 31;
  const int wave = tid >> 5;
  const int wm = (wave & 3) * 32;
  const int wn = (wave >> 2) * 32;
  const int h0 = blockIdx.y * 64;

  if (tid < 128) {
    int pp = r0 + tid; if (pp > ce - 1) pp = ce - 1;
    sSlot[tid] = idxSlot[e * TOK + pp];
  }

  v8f ct[2][2];
#pragma unroll
  for (int i = 0; i < 2; ++i)
#pragma unroll
    for (int j = 0; j < 2; ++j) ct[i][j] = vzero();

  gemm_seg(Hg + (size_t)e * TOK * FD, FD,
           Wd + (size_t)e * HD * FD, FD, FD,
           sA, sB, r0, h0, tid, lane, wm, wn, ct);

  const int rlo = (lane < 16) ? 0 : 8;
  const int cn  = lane & 15;
#pragma unroll
  for (int i = 0; i < 2; ++i)
#pragma unroll
    for (int j = 0; j < 2; ++j)
#pragma unroll
      for (int r = 0; r < 8; ++r) {
        const int lr = wm + i * 16 + r + rlo;
        if (r0 + lr < ce) {
          const int slot = sSlot[lr];
          const int h = h0 + wn + j * 16 + cn;
          Dslot[(size_t)slot * HD + h] = (_Float16)ct[i][j][r];
        }
      }
}

// -------- shared-expert down GEMM, writes base output --------
__global__ void __launch_bounds__(256) downshared_kernel(const _Float16* __restrict__ Hs,
                                                         const float* __restrict__ SWd,
                                                         float* __restrict__ out) {
  __shared__ __align__(16) _Float16 sA[2][128 * PITCH];
  __shared__ __align__(16) _Float16 sB[2][64 * PITCH];
  const int tid  = threadIdx.x;
  const int lane = tid & 31;
  const int wave = tid >> 5;
  const int wm = (wave & 3) * 32;
  const int wn = (wave >> 2) * 32;
  const int t0 = blockIdx.x * 128;
  const int h0 = blockIdx.y * 64;

  v8f cf[2][2];
#pragma unroll
  for (int i = 0; i < 2; ++i)
#pragma unroll
    for (int j = 0; j < 2; ++j) cf[i][j] = vzero();

  gemm_seg(Hs, FSD, SWd, FSD, FSD, sA, sB, t0, h0, tid, lane, wm, wn, cf);

  const int rlo = (lane < 16) ? 0 : 8;
  const int cn  = lane & 15;
#pragma unroll
  for (int i = 0; i < 2; ++i)
#pragma unroll
    for (int j = 0; j < 2; ++j)
#pragma unroll
      for (int r = 0; r < 8; ++r) {
        const int t = t0 + wm + i * 16 + r + rlo;
        const int h = h0 + wn + j * 16 + cn;
        out[(size_t)t * HD + h] = cf[i][j][r];
      }
}

// -------- final combine: out += w0*Dslot[t,0] + w1*Dslot[t,1] --------
__global__ void __launch_bounds__(256) combine_kernel(const _Float16* __restrict__ Dslot,
                                                      const float* __restrict__ w,
                                                      float* __restrict__ out) {
  const int gid = blockIdx.x * 256 + threadIdx.x;     // T * (HD/4) threads
  const int t  = gid >> 9;                            // HD/4 = 512 chunks/token
  const int hc = (gid & 511) * 4;
  const float w0 = w[t * KK + 0];
  const float w1 = w[t * KK + 1];
  const _Float16* d0 = Dslot + (size_t)(t * KK + 0) * HD + hc;
  const _Float16* d1 = d0 + HD;
  float4* op = (float4*)(out + (size_t)t * HD + hc);
  float4 o = *op;
  o.x += w0 * (float)d0[0] + w1 * (float)d1[0];
  o.y += w0 * (float)d0[1] + w1 * (float)d1[1];
  o.z += w0 * (float)d0[2] + w1 * (float)d1[2];
  o.w += w0 * (float)d0[3] + w1 * (float)d1[3];
  *op = o;
}

extern "C" void kernel_launch(void* const* d_in, const int* in_sizes, int n_in,
                              void* d_out, int out_size, void* d_ws, size_t ws_size,
                              hipStream_t stream) {
  (void)in_sizes; (void)n_in; (void)out_size; (void)ws_size;
  const float* X   = (const float*)d_in[0];  // [B,S,H]
  const float* GW  = (const float*)d_in[1];  // [E,H]
  const float* Wg  = (const float*)d_in[2];  // [E,F,H]
  const float* Wu  = (const float*)d_in[3];  // [E,F,H]
  const float* Wd  = (const float*)d_in[4];  // [E,H,F]
  const float* SWg = (const float*)d_in[5];  // [FS,H]
  const float* SWu = (const float*)d_in[6];  // [FS,H]
  const float* SWd = (const float*)d_in[7];  // [H,FS]
  float* out = (float*)d_out;

  char* ws = (char*)d_ws;
  int*      cnt     = (int*)ws;                          // 256 B (8 used)
  float*    w       = (float*)(ws + 4096);               // T*K*4   = 16 KB
  int*      idxSlot = (int*)(ws + 32768);                // E*T*4   = 64 KB
  _Float16* Xh      = (_Float16*)(ws + 131072);          // T*H*2   = 8 MB
  _Float16* Hg      = (_Float16*)(ws + 131072 + (size_t)TOK * HD * 2);        // E*T*F*2 = 46 MB
  _Float16* Hs      = (_Float16*)((char*)Hg + (size_t)NE * TOK * FD * 2);     // T*FS*2  = 11.5 MB
  _Float16* Dslot   = (_Float16*)((char*)Hs + (size_t)TOK * FSD * 2);         // T*K*H*2 = 16 MB

  hipMemsetAsync(cnt, 0, 256, stream);
  gate_kernel<<<TOK / 8, 256, 0, stream>>>(X, GW, Xh, w, cnt, idxSlot);
  upshared_kernel<<<dim3(TOK / 128, FSD / 64), 256, 0, stream>>>(Xh, SWg, SWu, Hs);
  upgather_kernel<<<dim3(TOK / 128, FD / 64, NE), 256, 0, stream>>>(Xh, Wg, Wu, Hg, cnt, idxSlot);
  downgather_kernel<<<dim3(TOK / 128, HD / 64, NE), 256, 0, stream>>>(Hg, Wd, Dslot, cnt, idxSlot);
  downshared_kernel<<<dim3(TOK / 128, HD / 64), 256, 0, stream>>>(Hs, SWd, out);
  combine_kernel<<<(TOK * (HD / 4)) / 256, 256, 0, stream>>>(Dslot, w, out);
}
